// mHC_9809705304399
// MI455X (gfx1250) — matrix operand order repaired and checked by reading, not compiled, not run on hardware
//
#include <hip/hip_runtime.h>
#include <hip/hip_bf16.h>
#include <math.h>

// ---------------------------------------------------------------------------
// Problem dims (B=2, S=4096, N=4, D=2048)
// ---------------------------------------------------------------------------
#define T_TOKENS 8192   // B*S
#define N_ARMS   4
#define DDIM     2048
#define KDIM     8192   // N*D
#define NOUT     24     // 2N + N*N
#define NPAD     32     // padded output columns
#define WROW     64     // halves per packed W row: [0..31]=hi, [32..63]=lo
#define KCHUNK   256    // K rows staged in LDS per block iteration

typedef __attribute__((ext_vector_type(16))) __bf16 v16bf;
typedef __attribute__((ext_vector_type(8)))  __bf16 v8bf;
typedef __attribute__((ext_vector_type(8)))  float  v8f;

// ---------------------------------------------------------------------------
// K0: pack hc_weight (8192x24 f32) into bf16 (hi, lo) pair, padded to 32 cols
// ---------------------------------------------------------------------------
__global__ void mhc_pack_w(const float* __restrict__ w, __bf16* __restrict__ wb) {
    int idx = blockIdx.x * blockDim.x + threadIdx.x;   // over KDIM*NPAD
    if (idx >= KDIM * NPAD) return;
    int k = idx / NPAD, c = idx % NPAD;
    float f  = (c < NOUT) ? w[k * NOUT + c] : 0.0f;
    __bf16 hi = (__bf16)f;
    __bf16 lo = (__bf16)(f - (float)hi);
    wb[(size_t)k * WROW + c]      = hi;
    wb[(size_t)k * WROW + 32 + c] = lo;
}

// ---------------------------------------------------------------------------
// K1: WMMA GEMM  dot[T][32] = X(bf16) @ (W_hi + W_lo),  + per-token sum(x^2)
//     1 wave = one 16-token M-tile; block = 4 waves; W chunk staged in LDS.
// ---------------------------------------------------------------------------
__global__ __launch_bounds__(128) void mhc_gemm(
    const __bf16* __restrict__ x, const __bf16* __restrict__ wb,
    float* __restrict__ dot, float* __restrict__ sumsq) {

    __shared__ __attribute__((aligned(32))) __bf16 sW[KCHUNK * WROW]; // 32 KB

    const int tid    = threadIdx.x;
    const int wave   = tid >> 5;
    const int lane   = tid & 31;
    const int laneLo = lane & 15;
    const int hiSel  = (lane >> 4) & 1;            // 0: lanes 0-15, 1: lanes 16-31
    const int m0     = (blockIdx.x * 4 + wave) * 16;

    v8f c0 = {}; v8f c1 = {};
    float ss = 0.0f;

    for (int kc = 0; kc < KDIM; kc += KCHUNK) {
        __syncthreads();
        // cooperative load: 256 rows x 128 B = 32 KB, 128 threads x 16 B x 16
        const int4* gsrc = (const int4*)(wb + (size_t)kc * WROW);
        int4* ldst = (int4*)sW;
        #pragma unroll
        for (int i = 0; i < 16; ++i)
            ldst[tid + i * 128] = gsrc[tid + i * 128];
        __syncthreads();

        #pragma unroll
        for (int j = 0; j < KCHUNK / 32; ++j) {
            const int kg = kc + j * 32;
            // --- A fragment (ISA 16-bit A 16x32 layout) ---
            // lane L: row M = L&15; halves 0..7 -> K = kg + hiSel*8 + h
            //                      halves 8..15 -> K = kg + 16 + hiSel*8 + h
            const __bf16* arow = x + (size_t)(m0 + laneLo) * KDIM + kg + hiSel * 8;
            union { v16bf v; v8bf h[2]; } a;
            a.h[0] = *(const v8bf*)(arow);
            a.h[1] = *(const v8bf*)(arow + 16);
            // --- per-row sum of squares (exact f32, x is exact in bf16) ---
            #pragma unroll
            for (int e = 0; e < 8; ++e) {
                float f0 = (float)a.h[0][e]; ss = fmaf(f0, f0, ss);
                float f1 = (float)a.h[1][e]; ss = fmaf(f1, f1, ss);
            }
            // --- B fragments (lane L: column N = L&15; halves use same K order as A) ---
            v16bf b0hi, b1hi, b0lo, b1lo;
            #pragma unroll
            for (int s = 0; s < 16; ++s) {
                const int kk = (s < 8) ? (hiSel * 8 + s) : (16 + hiSel * 8 + (s - 8));
                const __bf16* bp = sW + (size_t)(j * 32 + kk) * WROW + laneLo;
                b0hi[s] = bp[0]; b1hi[s] = bp[16]; b0lo[s] = bp[32]; b1lo[s] = bp[48];
            }
            c0 = __builtin_amdgcn_wmma_f32_16x16x32_bf16(false, a.v, false, b0hi, (short)0, c0, false, false);
            c0 = __builtin_amdgcn_wmma_f32_16x16x32_bf16(false, a.v, false, b0lo, (short)0, c0, false, false);
            c1 = __builtin_amdgcn_wmma_f32_16x16x32_bf16(false, a.v, false, b1hi, (short)0, c1, false, false);
            c1 = __builtin_amdgcn_wmma_f32_16x16x32_bf16(false, a.v, false, b1lo, (short)0, c1, false, false);
        }
    }

    // lane L and L+16 hold complementary K-halves of row M = L&15
    float other = __shfl_xor(ss, 16, 32);
    ss += other;
    if (lane < 16) sumsq[m0 + lane] = ss;

    // C layout: VGPR v -> M=v (lanes 0-15) / M=v+8 (lanes 16-31); N = lane&15
    #pragma unroll
    for (int v = 0; v < 8; ++v) {
        int m = m0 + v + hiSel * 8;
        dot[(size_t)m * NPAD + laneLo]      = c0[v];
        dot[(size_t)m * NPAD + 16 + laneLo] = c1[v];
    }
}

// ---------------------------------------------------------------------------
// K2: per-token gates: rsqrt scale, sigmoids, 4x4 Sinkhorn (matches reference)
// ---------------------------------------------------------------------------
__global__ void mhc_gates(const float* __restrict__ dot, const float* __restrict__ sumsq,
                          const float* __restrict__ alpha, const float* __restrict__ bias,
                          float* __restrict__ gates) {
    int t = blockIdx.x * blockDim.x + threadIdx.x;
    if (t >= T_TOKENS) return;
    const float EPS = 1e-6f;
    float rs = rsqrtf(sumsq[t] * (1.0f / (float)KDIM) + EPS);
    float xh[NOUT];
    #pragma unroll
    for (int c = 0; c < NOUT; ++c) xh[c] = dot[(size_t)t * NPAD + c] * rs;

    float a0 = alpha[0], a1 = alpha[1], a2 = alpha[2];
    float b0 = bias[0],  b1 = bias[1],  b2 = bias[2];

    float g[NOUT];
    #pragma unroll
    for (int n = 0; n < 4; ++n) {
        g[n]     = 1.0f / (1.0f + expf(-(a0 * xh[n] + b0))) + EPS;     // H_pre
        g[4 + n] = 2.0f / (1.0f + expf(-(a1 * xh[4 + n] + b1)));       // H_post
    }

    // Sinkhorn on H_comb_before = a2 * X_comb.reshape(4,4) + b2
    float H[16];
    #pragma unroll
    for (int m = 0; m < 4; ++m) {                       // row softmax + eps
        float r[4], mx = -3.4e38f;
        #pragma unroll
        for (int n = 0; n < 4; ++n) { r[n] = a2 * xh[8 + m * 4 + n] + b2; mx = fmaxf(mx, r[n]); }
        float s = 0.0f;
        #pragma unroll
        for (int n = 0; n < 4; ++n) { r[n] = expf(r[n] - mx); s += r[n]; }
        float inv = 1.0f / s;
        #pragma unroll
        for (int n = 0; n < 4; ++n) H[m * 4 + n] = r[n] * inv + EPS;
    }
    #pragma unroll
    for (int n = 0; n < 4; ++n) {                       // initial column norm
        float s = H[n] + H[4 + n] + H[8 + n] + H[12 + n];
        float inv = 1.0f / (s + EPS);
        #pragma unroll
        for (int m = 0; m < 4; ++m) H[m * 4 + n] *= inv;
    }
    for (int it = 0; it < 19; ++it) {                   // SINKHORN_ITERS-1
        #pragma unroll
        for (int m = 0; m < 4; ++m) {
            float s = H[m * 4] + H[m * 4 + 1] + H[m * 4 + 2] + H[m * 4 + 3];
            float inv = 1.0f / (s + EPS);
            #pragma unroll
            for (int n = 0; n < 4; ++n) H[m * 4 + n] *= inv;
        }
        #pragma unroll
        for (int n = 0; n < 4; ++n) {
            float s = H[n] + H[4 + n] + H[8 + n] + H[12 + n];
            float inv = 1.0f / (s + EPS);
            #pragma unroll
            for (int m = 0; m < 4; ++m) H[m * 4 + n] *= inv;
        }
    }
    #pragma unroll
    for (int i = 0; i < 16; ++i) g[8 + i] = H[i];

    #pragma unroll
    for (int c = 0; c < NOUT; ++c) gates[(size_t)t * NOUT + c] = g[c];
}

// ---------------------------------------------------------------------------
// K3: streaming output. 1 block = 1 token; thread covers 8 d-elements.
//     y[n,d] = H_post[n]*2*bf16(h_in)[d] + sum_m H_comb[m,n]*x[m,d]
// ---------------------------------------------------------------------------
__global__ __launch_bounds__(256) void mhc_output(
    const __bf16* __restrict__ x, const float* __restrict__ gates,
    __bf16* __restrict__ y) {
    const int t = blockIdx.x;
    const float* g = gates + (size_t)t * NOUT;
    float hpre[4], hpost[4], hc[16];
    #pragma unroll
    for (int n = 0; n < 4; ++n) { hpre[n] = g[n]; hpost[n] = g[4 + n]; }
    #pragma unroll
    for (int i = 0; i < 16; ++i) hc[i] = g[8 + i];

    const int d0 = threadIdx.x * 8;                    // 256*8 == 2048
    const __bf16* xr = x + (size_t)t * KDIM;
    v8bf xa[4];
    #pragma unroll
    for (int n = 0; n < 4; ++n) xa[n] = *(const v8bf*)(xr + n * DDIM + d0);

    v8bf yo[4];
    #pragma unroll
    for (int e = 0; e < 8; ++e) {
        float xf[4];
        #pragma unroll
        for (int n = 0; n < 4; ++n) xf[n] = (float)xa[n][e];
        float hin = 0.0f;
        #pragma unroll
        for (int n = 0; n < 4; ++n) hin = fmaf(hpre[n], xf[n], hin);
        // reference: h_in -> bf16, h_out = 2*h_in (bf16), then back to f32
        float hout = 2.0f * (float)((__bf16)hin);
        #pragma unroll
        for (int n = 0; n < 4; ++n) {
            float acc = hpost[n] * hout;
            #pragma unroll
            for (int m = 0; m < 4; ++m) acc = fmaf(hc[m * 4 + n], xf[m], acc);
            yo[n][e] = (__bf16)acc;
        }
    }
    __bf16* yr = y + (size_t)t * KDIM;
    #pragma unroll
    for (int n = 0; n < 4; ++n) *(v8bf*)(yr + n * DDIM + d0) = yo[n];
}

// ---------------------------------------------------------------------------
extern "C" void kernel_launch(void* const* d_in, const int* in_sizes, int n_in,
                              void* d_out, int out_size, void* d_ws, size_t ws_size,
                              hipStream_t stream) {
    const __bf16* x      = (const __bf16*)d_in[0];   // (2,4096,4,2048) bf16
    const float*  w      = (const float*)d_in[1];    // (8192,24) f32
    const float*  alpha  = (const float*)d_in[2];    // (3,)
    const float*  bias   = (const float*)d_in[3];    // (3,)
    __bf16*       y      = (__bf16*)d_out;

    // Workspace layout (~2.9 MB total)
    char* ws = (char*)d_ws;
    __bf16* wb    = (__bf16*)(ws);                                   // 1 MB
    float*  dot   = (float*)(ws + (size_t)KDIM * WROW * 2);          // 1 MB
    float*  sums  = (float*)(ws + (size_t)KDIM * WROW * 2
                                 + (size_t)T_TOKENS * NPAD * 4);     // 32 KB
    float*  gates = (float*)(ws + (size_t)KDIM * WROW * 2
                                 + (size_t)T_TOKENS * NPAD * 4
                                 + (size_t)T_TOKENS * 4);            // 768 KB

    mhc_pack_w<<<(KDIM * NPAD + 255) / 256, 256, 0, stream>>>(w, wb);
    mhc_gemm<<<T_TOKENS / (16 * 4), 128, 0, stream>>>(x, wb, dot, sums);
    mhc_gates<<<T_TOKENS / 256, 256, 0, stream>>>(dot, sums, alpha, bias, gates);
    mhc_output<<<T_TOKENS, 256, 0, stream>>>(x, gates, y);
}
